// TriangleAttentionEndingNode_86019605004399
// MI455X (gfx1250) — matrix operand order repaired and checked by reading, not compiled, not run on hardware
//
#include <hip/hip_runtime.h>

// ---------------------------------------------------------------------------
// TriangleAttentionEndingNode for MI455X (gfx1250, wave32, WMMA f32_16x16x32_f16)
// L=256, H=4, C=32, C_Z=128.  fp32 LN/softmax/accum, f16 matrix operands.
// Round 4: k1 re-tiled so blocks own a fixed j-column -> V/G tile results are
// contiguous in the 'a' axis and stored as one packed v8h (b128) per lane;
// store-phase branch hoisted out of the element loop (uniform, straight-line).
// ---------------------------------------------------------------------------

typedef __attribute__((ext_vector_type(16))) _Float16 v16h;
typedef __attribute__((ext_vector_type(8)))  _Float16 v8h;
typedef __attribute__((ext_vector_type(8)))  float    v8f;

#define Lr 256
#define Hh 4
#define Cc 32
#define CZ 128
#define SCALE 0.17677669529663687f   // 1/sqrt(32)

static __device__ __forceinline__ v16h cat16(v8h lo, v8h hi) {
  return __builtin_shufflevector(lo, hi, 0,1,2,3,4,5,6,7,8,9,10,11,12,13,14,15);
}
static __device__ __forceinline__ v8f wmma16(v16h a, v16h b, v8f c) {
  // D = A(16x32 f16) x B(32x16 f16) + C(16x16 f32)
  return __builtin_amdgcn_wmma_f32_16x16x32_f16(false, a, false, b, (short)0, c, false, false);
}
static __device__ __forceinline__ v8f vzero8() {
  v8f z = {0.f,0.f,0.f,0.f,0.f,0.f,0.f,0.f};
  return z;
}
static __device__ __forceinline__ float fast_rcp(float x) {
  return __builtin_amdgcn_rcpf(x);        // v_rcp_f32 (1 ulp)
}
static __device__ __forceinline__ float fast_sigmoid(float x) {
  return fast_rcp(1.0f + __expf(-x));
}

// ---------------------------------------------------------------------------
// Kernel 0: one-time weight prep.
//   Wt  f16 [512][128] : Wt[col][k] = concat(Wq|Wk|Wv|Wg)[k][col%128]
//   Wot f16 [128][128] : Wot[col][k] = Wo[k][col]
// ---------------------------------------------------------------------------
__global__ __launch_bounds__(256) void k0_prep(
    const float* __restrict__ Wq, const float* __restrict__ Wk,
    const float* __restrict__ Wv, const float* __restrict__ Wg,
    const float* __restrict__ Wo,
    _Float16* __restrict__ Wt, _Float16* __restrict__ Wot)
{
  const int tid = blockIdx.x * 256 + threadIdx.x;   // 0 .. 81919
  if (tid < 512 * 128) {
    const int col = tid >> 7, k = tid & 127;
    const int mat = col >> 7, cm = col & 127;
    const float* Wm = (mat == 0) ? Wq : (mat == 1) ? Wk : (mat == 2) ? Wv : Wg;
    Wt[tid] = (_Float16)Wm[k * 128 + cm];
  } else {
    const int t2 = tid - 512 * 128;                 // 0 .. 16383
    const int col = t2 >> 7, k = t2 & 127;
    Wot[t2] = (_Float16)Wo[k * 128 + col];
  }
}

// ---------------------------------------------------------------------------
// Kernel 1: fused LayerNorm + projections.
//   Per block: 16 rows of z with FIXED b (=j) and a in [a0, a0+16).
//   GEMM [16 x 128] x [128 x 512] over transposed f16 weights.
//   Q,K  -> [j][h][idx][c] f16 (8 x b16 stores/lane/tile)
//   V,G  -> transposed [j][h][c][idx] f16 (1 packed v8h store/lane/tile)
//   B    -> [h][k=a][i=b] f32
// ---------------------------------------------------------------------------
__global__ __launch_bounds__(128) void k1_ln_proj(
    const float* __restrict__ z, const float* __restrict__ gamma,
    const float* __restrict__ beta, const float* __restrict__ Wb,
    const _Float16* __restrict__ Wt,
    _Float16* __restrict__ Qg, _Float16* __restrict__ Kg,
    _Float16* __restrict__ Vt, _Float16* __restrict__ Gt,
    float* __restrict__ Bt)
{
  const int t       = threadIdx.x;
  const int lane    = t & 31;
  const int w       = t >> 5;
  const int half_id = lane >> 4;
  const int lm      = lane & 15;
  const int b_      = blockIdx.x & 255;          // j index (fixed per block)
  const int a0      = (blockIdx.x >> 8) << 4;    // first a (i/k) row

  __shared__ __align__(16) float    zs[16][128];   // raw z rows (fp32)
  __shared__ __align__(16) _Float16 zh[16][136];   // LN'd rows (f16, padded)

  // cooperative vectorized load of 16 rows z[a0+ri][b_][0:128]
  {
#pragma unroll
    for (int i = 0; i < 4; ++i) {
      const int idx  = t + i * 128;      // 0..511 float4s
      const int ri   = idx >> 5;
      const int col4 = idx & 31;
      ((float4*)&zs[ri][0])[col4] =
          ((const float4*)(z + ((size_t)(a0 + ri) * Lr + b_) * CZ))[col4];
    }
  }
  __syncthreads();

  // LayerNorm: 8 lanes per row, shfl reductions within lane-groups of 8
  {
    const int row  = t >> 3;
    const int part = t & 7;
    float s = 0.f;
#pragma unroll
    for (int u = 0; u < 16; ++u) s += zs[row][part * 16 + u];
    s += __shfl_xor(s, 1); s += __shfl_xor(s, 2); s += __shfl_xor(s, 4);
    const float mu = s * (1.0f / 128.0f);
    float v = 0.f;
#pragma unroll
    for (int u = 0; u < 16; ++u) {
      float d = zs[row][part * 16 + u] - mu; v += d * d;
    }
    v += __shfl_xor(v, 1); v += __shfl_xor(v, 2); v += __shfl_xor(v, 4);
    const float rstd = rsqrtf(v * (1.0f / 128.0f) + 1e-5f);
#pragma unroll
    for (int u = 0; u < 16; ++u) {
      const int k = part * 16 + u;
      zh[row][k] = (_Float16)((zs[row][k] - mu) * rstd * gamma[k] + beta[k]);
    }
  }
  __syncthreads();

  // main GEMM: 32 column-tiles of 16 over [Wq|Wk|Wv|Wg]; wave w owns 8 tiles
  for (int tt = 0; tt < 8; ++tt) {
    const int colg0 = (w * 8 + tt) * 16;
    // wave-uniform tile descriptors -> scalar branches in the store phase
    const int mat = __builtin_amdgcn_readfirstlane(colg0 >> 7);   // 0..3
    const int cm0 = __builtin_amdgcn_readfirstlane(colg0 & 127);

    v8f acc = vzero8();
#pragma unroll
    for (int ks = 0; ks < 4; ++ks) {
      const v8h* ar = (const v8h*)&zh[lm][ks * 32 + half_id * 16];
      const v16h a = cat16(ar[0], ar[1]);
      const v8h* wp =
          (const v8h*)(Wt + (size_t)(colg0 + lm) * 128 + ks * 32 + half_id * 16);
      const v16h b = cat16(wp[0], wp[1]);
      acc = wmma16(a, b, acc);
    }

    const int hh  = cm0 >> 5;                // head (uniform per tile)
    const int c   = (cm0 & 31) + lm;         // 0..31
    const size_t jhbase = (size_t)(b_ * Hh + hh) * (Lr * Cc);

    if (mat < 2) {
      // row-major [j][h][idx=a][c]; idx varies with r -> stride Cc halfs
      _Float16* dst = ((mat == 0) ? Qg : Kg) + jhbase + c;
      const int aa0 = a0 + 8 * half_id;
#pragma unroll
      for (int r = 0; r < 8; ++r)
        dst[(size_t)(aa0 + r) * Cc] = (_Float16)acc[r];
    } else {
      // transposed [j][h][c][idx=a]; idx varies with r -> contiguous v8h
      _Float16* dst = ((mat == 2) ? Vt : Gt) + jhbase + (size_t)c * Lr +
                      a0 + 8 * half_id;
      v8h vh;
      if (mat == 3) {
#pragma unroll
        for (int r = 0; r < 8; ++r) vh[r] = (_Float16)fast_sigmoid(acc[r]);
      } else {
#pragma unroll
        for (int r = 0; r < 8; ++r) vh[r] = (_Float16)acc[r];
      }
      *(v8h*)dst = vh;
    }
  }

  // bias projection B = z_ln @ Wb  (4 cols) -> Bt[h][k=a][i=b]
  if (t < 64) {
    const int row = t >> 2, hc = t & 3;
    float acc = 0.f;
#pragma unroll 8
    for (int k = 0; k < 128; ++k) acc += (float)zh[row][k] * Wb[k * Hh + hc];
    Bt[(size_t)hc * (Lr * Lr) + (size_t)(a0 + row) * Lr + b_] = acc;
  }
}

// ---------------------------------------------------------------------------
// Kernel 2: attention per (j,h).  1024 blocks x 128 threads (4 indep. waves).
//   S = Q K^T (one WMMA per 16x16 tile, K-dim = C = 32),
//   + scale + bias + res_mask, row softmax (shfl within 16-lane groups),
//   P (f16, via padded LDS strip) @ V, gate, store f16 [i][j][h*32+c].
// ---------------------------------------------------------------------------
__global__ __launch_bounds__(128) void k2_attn(
    const _Float16* __restrict__ Qg, const _Float16* __restrict__ Kg,
    const _Float16* __restrict__ Vt, const _Float16* __restrict__ Gt,
    const float* __restrict__ Bt, const unsigned char* __restrict__ resmask,
    _Float16* __restrict__ Og)
{
  const int t       = threadIdx.x;
  const int lane    = t & 31;
  const int w       = t >> 5;
  const int half_id = lane >> 4;
  const int lm      = lane & 15;
  const int jh      = blockIdx.x;          // j*H + h
  const int j       = jh >> 2;
  const int h       = jh & 3;

  const _Float16* Qb = Qg + (size_t)jh * Lr * Cc;
  const _Float16* Kb = Kg + (size_t)jh * Lr * Cc;
  const _Float16* Vb = Vt + (size_t)jh * Cc * Lr;
  const _Float16* Gb = Gt + (size_t)jh * Cc * Lr;
  const float*    Bb = Bt + (size_t)h * (Lr * Lr);   // [k][i]

  __shared__ __align__(16) _Float16 P[4][16][264];   // per-wave P strip, padded

  for (int itb = 0; itb < 4; ++itb) {
    const int it = w * 4 + itb;            // i-tile 0..15
    const int i0 = it * 16 + 8 * half_id;  // first of this lane-group's 8 rows

    // A-fragment of Q (rows it*16..+15, K-dim = c)
    const v8h* qr = (const v8h*)(Qb + (size_t)(it * 16 + lm) * Cc);
    const v16h aq = cat16(qr[2 * half_id], qr[2 * half_id + 1]);

    v8f S[16];
#pragma unroll
    for (int kt = 0; kt < 16; ++kt) S[kt] = vzero8();

#pragma unroll
    for (int kt = 0; kt < 16; ++kt) {
      const v8h* kr =
          (const v8h*)(Kb + (size_t)(kt * 16 + lm) * Cc + half_id * 16);
      S[kt] = wmma16(aq, cat16(kr[0], kr[1]), S[kt]);
    }

    // scale + bias + mask, track row max
    float rmax[8], rsum[8];
#pragma unroll
    for (int r = 0; r < 8; ++r) rmax[r] = -3.4e38f;

#pragma unroll
    for (int kt = 0; kt < 16; ++kt) {
      const int k = kt * 16 + lm;
      const bool ok = (resmask[k] != 0);
      const v8f bias8 = *(const v8f*)(Bb + (size_t)k * Lr + i0);  // 8 rows contig
#pragma unroll
      for (int r = 0; r < 8; ++r) {
        float s = ok ? fmaf(S[kt][r], SCALE, bias8[r]) : -__builtin_inff();
        S[kt][r] = s;
        rmax[r]  = fmaxf(rmax[r], s);
      }
    }
#pragma unroll
    for (int r = 0; r < 8; ++r) {
#pragma unroll
      for (int d = 1; d < 16; d <<= 1)
        rmax[r] = fmaxf(rmax[r], __shfl_xor(rmax[r], d));
      rsum[r] = 0.f;
    }
#pragma unroll
    for (int kt = 0; kt < 16; ++kt) {
#pragma unroll
      for (int r = 0; r < 8; ++r) {
        const float p = __expf(S[kt][r] - rmax[r]);
        S[kt][r] = p;
        rsum[r] += p;
      }
    }
#pragma unroll
    for (int r = 0; r < 8; ++r) {
#pragma unroll
      for (int d = 1; d < 16; d <<= 1) rsum[r] += __shfl_xor(rsum[r], d);
      rsum[r] = fast_rcp(rsum[r]);
    }

    // normalized probabilities -> per-wave LDS strip (D-layout -> A-layout)
#pragma unroll
    for (int kt = 0; kt < 16; ++kt)
#pragma unroll
      for (int r = 0; r < 8; ++r)
        P[w][r + 8 * half_id][kt * 16 + lm] = (_Float16)(S[kt][r] * rsum[r]);

    asm volatile("" ::: "memory");   // keep DS stores before DS loads (same wave)

    // O = P @ V  (K-dim = 256, 8 steps of 32; two c-tiles)
    v8f O0 = vzero8(), O1 = vzero8();
#pragma unroll
    for (int s8 = 0; s8 < 8; ++s8) {
      const v8h* pr = (const v8h*)&P[w][lm][s8 * 32 + half_id * 16];
      const v16h ap = cat16(pr[0], pr[1]);
      const v8h* v0 =
          (const v8h*)(Vb + (size_t)lm * Lr + s8 * 32 + half_id * 16);
      O0 = wmma16(ap, cat16(v0[0], v0[1]), O0);
      const v8h* v1 =
          (const v8h*)(Vb + (size_t)(16 + lm) * Lr + s8 * 32 + half_id * 16);
      O1 = wmma16(ap, cat16(v1[0], v1[1]), O1);
    }

    // gate with sigmoid(G) (transposed layout -> contiguous v8h per lane)
    const v8h g0 = *(const v8h*)(Gb + (size_t)lm * Lr + i0);
    const v8h g1 = *(const v8h*)(Gb + (size_t)(16 + lm) * Lr + i0);
#pragma unroll
    for (int r = 0; r < 8; ++r) {
      const int ig = it * 16 + r + 8 * half_id;
      const size_t o = ((size_t)ig * Lr + j) * (Hh * Cc) + h * Cc;
      Og[o + lm]      = (_Float16)(O0[r] * (float)g0[r]);
      Og[o + 16 + lm] = (_Float16)(O1[r] * (float)g1[r]);
    }
  }
}

// ---------------------------------------------------------------------------
// Kernel 3: output projection [65536 x 128] @ Wo[128 x 128], * pair_mask.
// ---------------------------------------------------------------------------
__global__ __launch_bounds__(128) void k3_out(
    const _Float16* __restrict__ Og, const _Float16* __restrict__ Wot,
    const float* __restrict__ pm, float* __restrict__ out)
{
  const int t       = threadIdx.x;
  const int lane    = t & 31;
  const int w       = t >> 5;
  const int half_id = lane >> 4;
  const int lm      = lane & 15;
  const int gr0     = blockIdx.x * 16;

  __shared__ __align__(16) _Float16 as[16][136];

  // stage 16 rows x 128 halfs (coalesced u32)
  {
    const unsigned int* src = (const unsigned int*)(Og + (size_t)gr0 * 128);
#pragma unroll
    for (int i = 0; i < 8; ++i) {
      const unsigned int u = src[t + i * 128];
      const int pos = (t + i * 128) * 2;
      *(unsigned int*)&as[pos >> 7][pos & 127] = u;
    }
  }
  __syncthreads();

#pragma unroll
  for (int nt2 = 0; nt2 < 2; ++nt2) {
    const int col = (w * 2 + nt2) * 16 + lm;
    v8f acc = vzero8();
#pragma unroll
    for (int ks = 0; ks < 4; ++ks) {
      const v8h* ar = (const v8h*)&as[lm][ks * 32 + half_id * 16];
      const v16h a = cat16(ar[0], ar[1]);
      const v8h* wp =
          (const v8h*)(Wot + (size_t)col * 128 + ks * 32 + half_id * 16);
      acc = wmma16(a, cat16(wp[0], wp[1]), acc);
    }
#pragma unroll
    for (int r = 0; r < 8; ++r) {
      const int gr = gr0 + r + 8 * half_id;
      out[(size_t)gr * 128 + col] = acc[r] * pm[gr];
    }
  }
}

// ---------------------------------------------------------------------------
extern "C" void kernel_launch(void* const* d_in, const int* in_sizes, int n_in,
                              void* d_out, int out_size, void* d_ws,
                              size_t ws_size, hipStream_t stream)
{
  const float*         z     = (const float*)d_in[0];
  const float*         pm    = (const float*)d_in[1];
  const unsigned char* rmask = (const unsigned char*)d_in[2];
  const float*         gamma = (const float*)d_in[3];
  const float*         beta  = (const float*)d_in[4];
  const float*         Wq    = (const float*)d_in[5];
  const float*         Wk    = (const float*)d_in[6];
  const float*         Wv    = (const float*)d_in[7];
  const float*         Wb    = (const float*)d_in[8];
  const float*         Wg    = (const float*)d_in[9];
  const float*         Wo    = (const float*)d_in[10];

  // workspace carve (~81.2 MB): 5 x 16MB f16 + 1MB f32 + f16 weights
  const size_t SZ = (size_t)Lr * Hh * Lr * Cc * sizeof(_Float16); // 16 MB
  char* p = (char*)d_ws;
  _Float16* Qg  = (_Float16*)p; p += SZ;
  _Float16* Kg  = (_Float16*)p; p += SZ;
  _Float16* Vt  = (_Float16*)p; p += SZ;
  _Float16* Gt  = (_Float16*)p; p += SZ;
  _Float16* Og  = (_Float16*)p; p += SZ;
  float*    Bt  = (float*)p;    p += (size_t)Hh * Lr * Lr * sizeof(float); // 1MB
  _Float16* Wt  = (_Float16*)p; p += (size_t)512 * 128 * sizeof(_Float16);
  _Float16* Wot = (_Float16*)p;

  k0_prep<<<dim3(320), dim3(256), 0, stream>>>(Wq, Wk, Wv, Wg, Wo, Wt, Wot);
  k1_ln_proj<<<dim3(4096), dim3(128), 0, stream>>>(
      z, gamma, beta, Wb, Wt, Qg, Kg, Vt, Gt, Bt);
  k2_attn<<<dim3(Lr * Hh), dim3(128), 0, stream>>>(
      Qg, Kg, Vt, Gt, Bt, rmask, Og);
  k3_out<<<dim3(4096), dim3(128), 0, stream>>>(
      Og, Wot, pm, (float*)d_out);
}
